// KimiDeltaAttention_5016521802514
// MI455X (gfx1250) — matrix operand order repaired and checked by reading, not compiled, not run on hardware
//
#include <hip/hip_runtime.h>

// ---------------- problem dims ----------------
static constexpr int B_  = 4;
static constexpr int T_  = 2048;
static constexpr int H_  = 16;
static constexpr int K_  = 128;
static constexpr int V_  = 128;
static constexpr int C_  = 64;    // chunk
static constexpr int NT_ = 32;    // T_/C_
static constexpr int VSPLIT = 2;  // independent V-column slices -> 2x workgroups
static constexpr int VB  = V_ / VSPLIT;   // 64
static constexpr int BDIM  = 256;         // 8 wave32
static constexpr int NWAVE = 8;

// padded LDS strides (elements) to avoid bank aliasing on 256B-multiples
static constexpr int KP = 136;   // bf16 rows of length K_
static constexpr int CP = 72;    // bf16 rows of length C_
static constexpr int SP = 132;   // f32 rows of length K_ (transposed state)

// ---------------- LDS layout (bytes), all 16B aligned ----------------
static constexpr size_t SM_STATET = 0;                         // f32  [VB][SP]   stateT[v][k]
static constexpr size_t SM_GCS    = SM_STATET + VB*SP*4;       // f32  [C_][K_]
static constexpr size_t SM_W      = SM_GCS    + C_*K_*4;       // f32  [C_][K_]
static constexpr size_t SM_U      = SM_W      + C_*K_*4;       // f32  [C_][VB]
static constexpr size_t SM_T      = SM_U      + C_*VB*4;       // f32  [C_][C_]
static constexpr size_t SM_GLAST  = SM_T      + C_*C_*4;       // f32  [K_]
static constexpr size_t SM_EGLAST = SM_GLAST  + K_*4;          // f32  [K_]  exp(g_last)
static constexpr size_t SM_BETA   = SM_EGLAST + K_*4;          // f32  [C_]
static constexpr size_t SM_STBT   = SM_BETA   + C_*4;          // bf16 [VB][KP]   state^T
static constexpr size_t SM_QG     = SM_STBT   + VB*KP*2;       // bf16 [C_][KP]
static constexpr size_t SM_KDEC   = SM_QG     + C_*KP*2;       // bf16 [C_][KP]
static constexpr size_t SM_TMP    = SM_KDEC   + C_*KP*2;       // bf16 [C_][KP] kgp, later vnewT [VB][CP]
static constexpr size_t SM_KGT    = SM_TMP    + C_*KP*2;       // bf16 [K_][CP]   kg^T[k][c]
static constexpr size_t SM_WBF    = SM_KGT    + K_*CP*2;       // bf16 [C_][KP]
static constexpr size_t SM_AQK    = SM_WBF    + C_*KP*2;       // bf16 [C_][CP]
static constexpr size_t SMEM_BYTES= SM_AQK    + C_*CP*2;       // ~242 KB < 320 KB

typedef __attribute__((ext_vector_type(16))) __bf16 v16bf;
typedef __attribute__((ext_vector_type(8)))  __bf16 v8bf;
typedef __attribute__((ext_vector_type(4)))  __bf16 v4bf;
typedef __attribute__((ext_vector_type(8)))  float  v8f;
typedef __attribute__((ext_vector_type(4)))  float  v4f;

// ---- WMMA fragment loaders (CDNA5 wave32 layouts, k = 8h+e, then 16+8h+e (AMD table order)) ----
__device__ __forceinline__ v16bf ldA(const __bf16* p, int stride, int r0, int k0, int lane) {
  const int m  = r0 + (lane & 15);
  const int kb = k0 + ((lane >> 4) << 3);
  const __bf16* q = p + m * stride + kb;
  v16bf a;
#pragma unroll
  for (int e = 0; e < 8; ++e) { a[e] = q[e]; a[e + 8] = q[e + 16]; }
  return a;
}
// B 32x16 from a transposed store: B[kk][n] = p[n0+n][k0+kk]  (same k order as ldA)
__device__ __forceinline__ v16bf ldB_T(const __bf16* p, int stride, int k0, int n0, int lane) {
  const int nn = n0 + (lane & 15);
  const int kb = k0 + ((lane >> 4) << 3);
  const __bf16* q = p + nn * stride + kb;
  v16bf b;
#pragma unroll
  for (int e = 0; e < 8; ++e) { b[e] = q[e]; b[e + 8] = q[e + 16]; }
  return b;
}
__device__ __forceinline__ v8f wmma_bf16(v16bf a, v16bf b, v8f c) {
  return __builtin_amdgcn_wmma_f32_16x16x32_bf16(false, a, false, b, (short)0, c, false, false);
}

__global__ __launch_bounds__(BDIM) void kda_fwd_kernel(
    const float* __restrict__ q, const float* __restrict__ k,
    const float* __restrict__ v, const float* __restrict__ g,
    const float* __restrict__ beta,
    float* __restrict__ out_o, float* __restrict__ out_state)
{
  extern __shared__ char smem[];
  float*  s_stateT = (float*)(smem + SM_STATET);
  float*  s_gcs    = (float*)(smem + SM_GCS);
  float*  s_w      = (float*)(smem + SM_W);
  float*  s_u      = (float*)(smem + SM_U);
  float*  s_T      = (float*)(smem + SM_T);
  float*  s_glast  = (float*)(smem + SM_GLAST);
  float*  s_eglast = (float*)(smem + SM_EGLAST);
  float*  s_beta   = (float*)(smem + SM_BETA);
  __bf16* s_stbT   = (__bf16*)(smem + SM_STBT);
  __bf16* s_qg     = (__bf16*)(smem + SM_QG);
  __bf16* s_kdec   = (__bf16*)(smem + SM_KDEC);
  __bf16* s_tmp    = (__bf16*)(smem + SM_TMP);   // kgp, then vnewT
  __bf16* s_kgT    = (__bf16*)(smem + SM_KGT);
  __bf16* s_wbf    = (__bf16*)(smem + SM_WBF);
  __bf16* s_Aqk    = (__bf16*)(smem + SM_AQK);

  const int tid  = threadIdx.x;
  const int wid  = tid >> 5;
  const int lane = tid & 31;
  const int tjm  = wid >> 1;          // this wave's N(V/col)-tile: 0..3
  const int half = wid & 1;           // row-tile group within the pair
  const int vblk = blockIdx.x % VSPLIT;
  const int bh   = blockIdx.x / VSPLIT;
  const int hh   = bh % H_;
  const int bb   = bh / H_;
  const int v0   = vblk * VB;
  const float scale = 0.08838834764831845f; // 1/sqrt(128)

  // zero running state (incl. padding)
  for (int i = tid; i < VB * SP; i += BDIM) s_stateT[i] = 0.f;
  for (int i = tid; i < VB * KP; i += BDIM) s_stbT[i] = (__bf16)0.f;
  __syncthreads();

  for (int n = 0; n < NT_; ++n) {
    const int t0 = n * C_;

    // ---- Step 1a: stage g chunk into LDS (coalesced float4) ----
    {
      const size_t gbase = ((size_t)(bb * T_ + t0) * H_ + hh) * K_;
      for (int i = tid; i < C_ * K_ / 4; i += BDIM) {
        const int c  = i >> 5;             // K_/4 == 32
        const int k4 = (i & 31) << 2;
        *(v4f*)(&s_gcs[c * K_ + k4]) = *(const v4f*)(g + gbase + (size_t)c * H_ * K_ + k4);
      }
    }
    if (tid < C_) s_beta[tid] = beta[(size_t)(bb * T_ + t0 + tid) * H_ + hh];
    // prefetch next chunk rows (one cacheline/thread; fire-and-forget -> global_prefetch_b8)
    if (n + 1 < NT_) {
      const int c = tid & (C_ - 1), which = tid >> 6;
      const float* p = (which == 0) ? q : (which == 1) ? k : (which == 2) ? g : v;
      const size_t base = ((size_t)(bb * T_ + t0 + C_ + c) * H_ + hh) * (size_t)K_;
      __builtin_prefetch(p + base + ((which == 3) ? v0 : 0), 0, 1);
    }
    __syncthreads();

    // ---- Step 1b: per-channel cumsum in LDS ----
    if (tid < K_) {
      float run = 0.f;
      for (int c = 0; c < C_; ++c) {
        run += s_gcs[c * K_ + tid];
        s_gcs[c * K_ + tid] = run;
      }
      s_glast[tid]  = run;
      s_eglast[tid] = __expf(run);
    }
    __syncthreads();

    // ---- Step 2: derived operand matrices (float4-vectorized) ----
    for (int i = (tid << 2); i < C_ * K_; i += (BDIM << 2)) {
      const int c = i >> 7, kk = i & (K_ - 1);
      const size_t base = ((size_t)(bb * T_ + t0 + c) * H_ + hh) * K_;
      const v4f qv = *(const v4f*)(q + base + kk);
      const v4f kv = *(const v4f*)(k + base + kk);
      const v4f gv = *(const v4f*)(&s_gcs[i]);
      const float be = s_beta[c];
      v4bf qgv, kdv, kpv;
      v4f  wv;
#pragma unroll
      for (int e = 0; e < 4; ++e) {
        const float eg  = __expf(gv[e]);
        const float kgp = kv[e] * eg;
        qgv[e] = (__bf16)(qv[e] * eg);
        kdv[e] = (__bf16)(kv[e] * __expf(-gv[e]));
        kpv[e] = (__bf16)kgp;
        wv[e]  = be * kgp;
        s_kgT[(kk + e) * CP + c] = (__bf16)(kv[e] * __expf(s_glast[kk + e] - gv[e]));
      }
      *(v4bf*)(&s_qg[c * KP + kk])   = qgv;
      *(v4bf*)(&s_kdec[c * KP + kk]) = kdv;
      *(v4bf*)(&s_tmp[c * KP + kk])  = kpv;   // kgp (A of Akk)
      *(v4f*)(&s_w[i])               = wv;    // rhs of w-solve
    }
    for (int i = (tid << 2); i < C_ * VB; i += (BDIM << 2)) {
      const int c = i >> 6, vv = i & (VB - 1);
      const size_t base = ((size_t)(bb * T_ + t0 + c) * H_ + hh) * V_;
      const v4f vvv = *(const v4f*)(v + base + v0 + vv);
      v4f uv;
#pragma unroll
      for (int e = 0; e < 4; ++e) uv[e] = s_beta[c] * vvv[e];
      *(v4f*)(&s_u[i]) = uv;                  // rhs of u-solve
    }
    __syncthreads();

    // ---- Step 3: Aqk + (T = I + Akk); one N-tile per wave, shared B frags, 4 chains ----
    {
      const int ti0 = 2 * half, ti1 = 2 * half + 1;
      v8f aq0 = {}, aq1 = {}, ak0 = {}, ak1 = {};
#pragma unroll
      for (int kb = 0; kb < K_ / 32; ++kb) {
        const v16bf bf = ldB_T(s_kdec, KP, kb * 32, tjm * 16, lane);   // (k_dec)^T
        aq0 = wmma_bf16(ldA(s_qg,  KP, ti0 * 16, kb * 32, lane), bf, aq0);
        aq1 = wmma_bf16(ldA(s_qg,  KP, ti1 * 16, kb * 32, lane), bf, aq1);
        ak0 = wmma_bf16(ldA(s_tmp, KP, ti0 * 16, kb * 32, lane), bf, ak0);
        ak1 = wmma_bf16(ldA(s_tmp, KP, ti1 * 16, kb * 32, lane), bf, ak1);
      }
      const int ncol = tjm * 16 + (lane & 15);
#pragma unroll
      for (int x = 0; x < 2; ++x) {
        const v8f& aq = x ? aq1 : aq0;
        const v8f& ak = x ? ak1 : ak0;
        const int mrow = (2 * half + x) * 16 + ((lane >> 4) << 3);
#pragma unroll
        for (int r = 0; r < 8; ++r) {
          const int m = mrow + r;
          const float bm = s_beta[m];                       // unconditional
          s_Aqk[m * CP + ncol] = (__bf16)((m >= ncol) ? aq[r] * scale : 0.f);
          float tv = (m > ncol) ? bm * ak[r] : 0.f;
          tv = (m == ncol) ? 1.f : tv;
          s_T[m * C_ + ncol] = tv;
        }
      }
    }
    __syncthreads();

    // ---- Step 4: unit-lower forward substitution, thread-per-column ----
    if (tid < K_) {                       // w columns (conflict-free: consecutive cols)
      const int col = tid;
      for (int i = 1; i < C_; ++i) {
        float acc = s_w[i * K_ + col];
        for (int j = 0; j < i; ++j) acc -= s_T[i * C_ + j] * s_w[j * K_ + col];
        s_w[i * K_ + col] = acc;
      }
    } else if (tid < K_ + VB) {           // u columns
      const int col = tid - K_;
      for (int i = 1; i < C_; ++i) {
        float acc = s_u[i * VB + col];
        for (int j = 0; j < i; ++j) acc -= s_T[i * C_ + j] * s_u[j * VB + col];
        s_u[i * VB + col] = acc;
      }
    }
    __syncthreads();
    for (int i = (tid << 2); i < C_ * K_; i += (BDIM << 2)) {
      const int c = i >> 7, kk = i & (K_ - 1);
      const v4f wv = *(const v4f*)(&s_w[i]);
      v4bf ov;
#pragma unroll
      for (int e = 0; e < 4; ++e) ov[e] = (__bf16)wv[e];
      *(v4bf*)(&s_wbf[c * KP + kk]) = ov;
    }
    __syncthreads();

    // ---- Step 5: v_new = u - w@state  AND  o_acc = qg@state  (shared state B frags, 4 chains) ----
    v8f acc_o0, acc_o1;
    {
      const int ti0 = 2 * half, ti1 = 2 * half + 1;
      v8f av0 = {}, av1 = {}, ao0 = {}, ao1 = {};
#pragma unroll
      for (int kb = 0; kb < K_ / 32; ++kb) {
        const v16bf bf = ldB_T(s_stbT, KP, kb * 32, tjm * 16, lane);
        av0 = wmma_bf16(ldA(s_wbf, KP, ti0 * 16, kb * 32, lane), bf, av0);
        av1 = wmma_bf16(ldA(s_wbf, KP, ti1 * 16, kb * 32, lane), bf, av1);
        ao0 = wmma_bf16(ldA(s_qg,  KP, ti0 * 16, kb * 32, lane), bf, ao0);
        ao1 = wmma_bf16(ldA(s_qg,  KP, ti1 * 16, kb * 32, lane), bf, ao1);
      }
      const int ncol = tjm * 16 + (lane & 15);
#pragma unroll
      for (int x = 0; x < 2; ++x) {
        const v8f& av = x ? av1 : av0;
        const int mrow = (2 * half + x) * 16 + ((lane >> 4) << 3);
        v8bf outv;
#pragma unroll
        for (int r = 0; r < 8; ++r)
          outv[r] = (__bf16)(s_u[(mrow + r) * VB + ncol] - av[r]);
        *(v8bf*)(&s_tmp[ncol * CP + mrow]) = outv;     // vnewT, single b128 store
      }
      acc_o0 = ao0; acc_o1 = ao1;
    }
    __syncthreads();

    // ---- Step 5b+6 merged: o = scale*o_acc + Aqk@v_new  and
    //      state = exp(g_last)(.)state + kg^T@v_new  (shared vnewT B frags, 6 chains) ----
    {
      const int ncol = tjm * 16 + (lane & 15);
      const v16bf bf0 = ldB_T(s_tmp, CP, 0,  tjm * 16, lane);
      const v16bf bf1 = ldB_T(s_tmp, CP, 32, tjm * 16, lane);
      // o tiles (rows over C_)
#pragma unroll
      for (int x = 0; x < 2; ++x) {
        const int ti = 2 * half + x;
        const v8f& ao = x ? acc_o1 : acc_o0;
        v8f acc2;
#pragma unroll
        for (int r = 0; r < 8; ++r) acc2[r] = ao[r] * scale;
        acc2 = wmma_bf16(ldA(s_Aqk, CP, ti * 16, 0,  lane), bf0, acc2);
        acc2 = wmma_bf16(ldA(s_Aqk, CP, ti * 16, 32, lane), bf1, acc2);
        const int mrow = ti * 16 + ((lane >> 4) << 3);
#pragma unroll
        for (int r = 0; r < 8; ++r)
          out_o[((size_t)(bb * T_ + t0 + mrow + r) * H_ + hh) * V_ + v0 + ncol] = acc2[r];
      }
      // state tiles (rows over K_)
#pragma unroll
      for (int x = 0; x < 4; ++x) {
        const int ti = 4 * half + x;
        v8f acc = {};
        acc = wmma_bf16(ldA(s_kgT, CP, ti * 16, 0,  lane), bf0, acc);
        acc = wmma_bf16(ldA(s_kgT, CP, ti * 16, 32, lane), bf1, acc);
        const int mrow = ti * 16 + ((lane >> 4) << 3);  // K index base
        float* sp = &s_stateT[ncol * SP + mrow];
        v4f o0 = *(v4f*)(sp);
        v4f o1 = *(v4f*)(sp + 4);
        v8bf nb;
#pragma unroll
        for (int r = 0; r < 4; ++r) {
          const float ns = o0[r] * s_eglast[mrow + r] + acc[r];
          o0[r] = ns; nb[r] = (__bf16)ns;
        }
#pragma unroll
        for (int r = 0; r < 4; ++r) {
          const float ns = o1[r] * s_eglast[mrow + 4 + r] + acc[4 + r];
          o1[r] = ns; nb[4 + r] = (__bf16)ns;
        }
        *(v4f*)(sp)     = o0;
        *(v4f*)(sp + 4) = o1;
        *(v8bf*)(&s_stbT[ncol * KP + mrow]) = nb;       // single b128 store
      }
    }
    __syncthreads();
  }

  // ---- final state -> global ----
  const size_t sbase = (size_t)B_ * T_ * H_ * V_ + (size_t)(bb * H_ + hh) * K_ * V_;
  for (int i = tid; i < K_ * VB; i += BDIM) {
    const int kk = i / VB, vv = i % VB;
    out_state[sbase + (size_t)kk * V_ + v0 + vv] = s_stateT[vv * SP + kk];
  }
}

extern "C" void kernel_launch(void* const* d_in, const int* in_sizes, int n_in,
                              void* d_out, int out_size, void* d_ws, size_t ws_size,
                              hipStream_t stream) {
  const float* q    = (const float*)d_in[0];
  const float* k    = (const float*)d_in[1];
  const float* v    = (const float*)d_in[2];
  const float* g    = (const float*)d_in[3];
  const float* beta = (const float*)d_in[4];
  float* out = (float*)d_out;
  float* out_state = out + (size_t)B_ * T_ * H_ * V_;

  dim3 grid(B_ * H_ * VSPLIT);   // 128 workgroups, one (b,h,v-slice) each
  dim3 block(BDIM);
  hipLaunchKernelGGL(kda_fwd_kernel, grid, block, SMEM_BYTES, stream,
                     q, k, v, g, beta, out, out_state);
}
